// DFN_54820962566841
// MI455X (gfx1250) — compile-verified, matrix operand order checked by reading, not run on hardware
//
#include <hip/hip_runtime.h>
#include <hip/hip_bf16.h>

typedef __attribute__((ext_vector_type(16))) _Float16 v16h;
typedef __attribute__((ext_vector_type(8)))  _Float16 v8h;
typedef __attribute__((ext_vector_type(8)))  float    v8f;

#define BATCH   4
#define TSTEPS  19
#define SLOPE   0.01f

// ---------------------------------------------------------------------------
// Swizzled weight packing (for WMMA convs, Cin >= 32).
// K is tap-major: k = tap*Cin + cin  (tap = ky*KS+kx).
// Output layout: [mTile][chunk][lane][16 halves] so the A fragment is one
// contiguous 32-byte load per lane. kpat(e,lane) is the ISA 16-bit A map.
// ---------------------------------------------------------------------------
__global__ void pack_weights_swz_kernel(const float* __restrict__ w,
                                        _Float16* __restrict__ wp,
                                        int Cout, int Cin, int KS2,
                                        int nCh, int n) {
    int idx = blockIdx.x * blockDim.x + threadIdx.x;
    if (idx >= n) return;
    int e    = idx & 15;
    int lane = (idx >> 4) & 31;
    int g    = idx >> 9;              // mTile*nCh + ch
    int ch   = g % nCh;
    int mTile = g / nCh;
    int kb   = (lane >> 4) * 8;
    int i    = e >> 1, j = e & 1;
    int kpat = (i < 4) ? (kb + 2 * i + j) : (16 + kb + 2 * (i - 4) + j);
    int kp   = ch * 32 + kpat;        // tap-major K index
    int m    = mTile * 16 + (lane & 15);
    int Ktot = Cin * KS2;
    float v = 0.0f;
    if (m < Cout && kp < Ktot) {
        int tap = kp / Cin;
        int cin = kp - tap * Cin;
        v = w[((size_t)m * Cin + cin) * KS2 + tap];    // OIHW row-major
    }
    wp[idx] = (_Float16)v;
}

__global__ void zero16_kernel(_Float16* p, int n) {
    int i = blockIdx.x * blockDim.x + threadIdx.x;
    if (i < n) p[i] = (_Float16)0.0f;
}

// ---------------------------------------------------------------------------
// Blend: a = m*frame + (1-m)*x_gen  (m==1 for t<10).
// Writes f32 plain [B][128][128] (for DFN) and f16 padded NHWC [B][130][130][1].
// ---------------------------------------------------------------------------
__global__ void blend_kernel(const float* __restrict__ frames,
                             const float* __restrict__ mask_true,
                             const float* __restrict__ prev_out,
                             float* __restrict__ a32,
                             _Float16* __restrict__ aP, int t) {
    int p = blockIdx.x * blockDim.x + threadIdx.x;
    if (p >= BATCH * 16384) return;
    int b = p >> 14, pix = p & 16383;
    int y = pix >> 7, x = pix & 127;
    float fr = frames[((size_t)b * 20 + t) * 16384 + pix];
    float v;
    if (t < 10) {
        v = fr;
    } else {
        float m  = mask_true[((size_t)b * 10 + (t - 10)) * 16384 + pix];
        float xg = prev_out[((size_t)b * TSTEPS + (t - 1)) * 16384 + pix];
        v = m * fr + (1.0f - m) * xg;
    }
    a32[p] = v;
    aP[((size_t)b * 130 + y + 1) * 130 + (x + 1)] = (_Float16)v;
}

// ---------------------------------------------------------------------------
// wx0: Cin=1 3x3 conv (~0.13% of FLOPs) as a direct VALU conv.
// in: padded [B][130][130][1]; out: padded NHWC [B][130][130][32].
// ---------------------------------------------------------------------------
__global__ void conv1_direct_kernel(const _Float16* __restrict__ aP,
                                    const float* __restrict__ w,
                                    const float* __restrict__ bias,
                                    _Float16* __restrict__ out) {
    int p = blockIdx.x * blockDim.x + threadIdx.x;
    if (p >= BATCH * 16384) return;
    int b = p >> 14, pix = p & 16383;
    int y = pix >> 7, x = pix & 127;
    const _Float16* ab = aP + (size_t)b * 130 * 130;
    float av[9];
#pragma unroll
    for (int tap = 0; tap < 9; ++tap)
        av[tap] = (float)ab[(size_t)(y + tap / 3) * 130 + (x + tap % 3)];
    _Float16* ob = out + (((size_t)b * 130 + y + 1) * 130 + (x + 1)) * 32;
#pragma unroll
    for (int m = 0; m < 32; ++m) {
        float acc = bias[m];
#pragma unroll
        for (int tap = 0; tap < 9; ++tap) acc += w[m * 9 + tap] * av[tap];
        acc = (acc > 0.0f) ? acc : SLOPE * acc;
        ob[m] = (_Float16)acc;
    }
}

// ---------------------------------------------------------------------------
// 2x nearest upsample, padded NHWC f16, C=64: [B][66][66][64] -> [B][130][130][64]
// ---------------------------------------------------------------------------
__global__ void upsample2_kernel(const _Float16* __restrict__ in,
                                 _Float16* __restrict__ out) {
    int idx = blockIdx.x * blockDim.x + threadIdx.x;
    if (idx >= (BATCH << 20)) return;            // 4*128*128*64
    int c = idx & 63;
    int x = (idx >> 6) & 127;
    int y = (idx >> 13) & 127;
    int b = idx >> 20;
    out[(((size_t)b * 130 + y + 1) * 130 + x + 1) * 64 + c] =
        in[(((size_t)b * 66 + (y >> 1) + 1) * 66 + (x >> 1) + 1) * 64 + c];
}

// ---------------------------------------------------------------------------
// Implicit-GEMM conv, WMMA f32 += f16 x f16 (16x16x32).
// Input:  padded NHWC f16 [B][HIN+2][WIN+2][CIN], zero halo -> NO bounds checks.
// A: swizzled weights -> one v16h load / chunk.  Tap-major K: each 32-chunk is
// one filter tap (CIN multiple of 32), so B frag = 2 contiguous b128 loads at
// a fixed pixel.  Each wave: 1 M-tile x 4 N-tiles (A reuse, hazard cover).
// C/D layout: VGPR r -> M = r + (lane<16?0:8), N = lane&15.
// ---------------------------------------------------------------------------
template <int CIN, int KS, int STRIDE, int HIN>
__global__ void conv_wmma_kernel(const _Float16* __restrict__ in,
                                 const _Float16* __restrict__ wp,
                                 const float* __restrict__ bias,
                                 const float* __restrict__ addsrc,  // unpadded NHWC f32
                                 _Float16* __restrict__ outP,       // padded NHWC f16
                                 _Float16* __restrict__ outU,       // unpadded NHWC f16
                                 float* __restrict__ out32,         // unpadded NHWC f32
                                 int Cout, int relu) {
    constexpr int WIN  = HIN;
    constexpr int HP   = HIN + 2, WP = WIN + 2;
    constexpr int HOUT = HIN / STRIDE, WOUT = WIN / STRIDE;
    constexpr int KS2  = KS * KS;
    constexpr int OFS  = 1 - KS / 2;              // padded-coord offset
    constexpr int NT   = 4;
    constexpr int nCh  = (CIN * KS2) / 32;        // exact: CIN % 32 == 0

    const int lane = threadIdx.x & 31;
    const int wave = threadIdx.x >> 5;
    const int tileBase = (blockIdx.x * 4 + wave) * NT;
    const int mBase    = blockIdx.y * 16;
    const int b        = blockIdx.z;
    const int l15  = lane & 15;
    const int half = lane >> 4;
    const int kb   = half * 8;

    int oyT[NT], oxL[NT];
    size_t baseOff[NT];
#pragma unroll
    for (int nt = 0; nt < NT; ++nt) {
        int nBase = (tileBase + nt) * 16;
        int oy = nBase / WOUT;                    // pow2 -> shift; tile-uniform
        int ox = (nBase % WOUT) + l15;
        oyT[nt] = oy;
        oxL[nt] = ox;
        baseOff[nt] = ((size_t)(oy * STRIDE + OFS) * WP + (ox * STRIDE + OFS)) * CIN;
    }

    const _Float16* __restrict__ inb = in + (size_t)b * HP * WP * CIN;
    const _Float16* __restrict__ wpt = wp + (size_t)blockIdx.y * nCh * 512 + lane * 16;

    v8f acc[NT] = {};
#pragma unroll 1
    for (int ch = 0; ch < nCh; ++ch) {
        v16h afrag = *(const v16h*)(wpt + (size_t)ch * 512);
        const int kc   = ch * 32;
        const int tap  = kc / CIN;                // chunk-uniform (CIN % 32 == 0)
        const int cinB = kc % CIN;
        const int dy = tap / KS, dx = tap % KS;
        const size_t tapOff = ((size_t)dy * WP + dx) * CIN + cinB + kb;
#pragma unroll
        for (int nt = 0; nt < NT; ++nt) {
            const _Float16* bp = inb + baseOff[nt] + tapOff;
            v8h r0 = *(const v8h*)bp;
            v8h r1 = *(const v8h*)(bp + 16);
            v16h bfrag = __builtin_shufflevector(r0, r1,
                0, 1, 2, 3, 4, 5, 6, 7, 8, 9, 10, 11, 12, 13, 14, 15);
            acc[nt] = __builtin_amdgcn_wmma_f32_16x16x32_f16(
                false, afrag, false, bfrag, (short)0, acc[nt], false, false);
        }
    }

#pragma unroll
    for (int nt = 0; nt < NT; ++nt) {
        int oy = oyT[nt], ox = oxL[nt];
#pragma unroll
        for (int r = 0; r < 8; ++r) {
            int m = mBase + r + half * 8;
            if (m < Cout) {
                float v = acc[nt][r] + bias[m];
                if (relu) v = (v > 0.0f) ? v : SLOPE * v;
                size_t iu = (((size_t)b * HOUT + oy) * WOUT + ox) * Cout + m;
                if (addsrc) v += addsrc[iu];
                if (out32)  out32[iu] = v;
                if (outU)   outU[iu] = (_Float16)v;
                if (outP)
                    outP[(((size_t)b * (HOUT + 2) + oy + 1) * (WOUT + 2) + ox + 1) * Cout + m]
                        = (_Float16)v;
            }
        }
    }
}

// ---------------------------------------------------------------------------
// Softmax over 81 filter logits + 9x9 dynamic local filtering.
// logits: unpadded NHWC f16 stride 82 (contiguous per pixel).
// ---------------------------------------------------------------------------
__global__ void dfn_kernel(const _Float16* __restrict__ logits,
                           const float* __restrict__ a32,
                           float* __restrict__ out, int t) {
    int p = blockIdx.x * blockDim.x + threadIdx.x;
    if (p >= BATCH * 16384) return;
    int b = p >> 14, pix = p & 16383;
    int y = pix >> 7, x = pix & 127;
    const _Float16* lg = logits + (size_t)p * 82;
    const float*    ab = a32 + (size_t)b * 16384;

    float mx = -1e30f;
    for (int c = 0; c < 81; ++c) mx = fmaxf(mx, (float)lg[c]);
    float s = 0.0f, acc = 0.0f;
    for (int c = 0; c < 81; ++c) {
        float e = __expf((float)lg[c] - mx);
        s += e;
        int dy = c / 9 - 4, dx = c % 9 - 4;
        int yy = y + dy, xx = x + dx;
        float av = (yy >= 0 && yy < 128 && xx >= 0 && xx < 128)
                       ? ab[yy * 128 + xx] : 0.0f;
        acc += e * av;
    }
    out[((size_t)b * TSTEPS + t) * 16384 + pix] = acc / s;
}

// ---------------------------------------------------------------------------
// Host orchestration
// ---------------------------------------------------------------------------
struct ConvDesc { int Cout, Cin, ks; };
// order: wx0..wx4, ws0, ws1, wo0..wo4, last
static const ConvDesc g_convs[13] = {
    {32, 1, 3}, {32, 32, 3}, {64, 32, 3}, {64, 64, 3}, {128, 64, 3},
    {128, 128, 3}, {128, 128, 3},
    {64, 128, 3}, {64, 64, 3}, {64, 64, 3}, {64, 64, 3}, {128, 64, 1},
    {82, 128, 1}
};

static inline size_t align256(size_t x) { return (x + 255) & ~(size_t)255; }

extern "C" void kernel_launch(void* const* d_in, const int* in_sizes, int n_in,
                              void* d_out, int out_size, void* d_ws, size_t ws_size,
                              hipStream_t stream) {
    (void)n_in; (void)out_size; (void)ws_size;

    const float* frames    = (const float*)d_in[0];
    const float* mask_true = (const float*)d_in[1];

    // Map param tensors (two possible pytree flattenings of 'params').
    const float* W[13];
    const float* Bv[13];
    if (in_sizes[2] == 288) {
        int i = 2;   // insertion order: wx(5), ws(2), wo(5), last
        for (int l = 0; l < 13; ++l) { W[l] = (const float*)d_in[i++]; Bv[l] = (const float*)d_in[i++]; }
    } else {
        static const int map[13] = {12, 7, 8, 9, 10, 11, 5, 6, 0, 1, 2, 3, 4};
        int i = 2;   // jax sorted keys: last, wo(5), ws(2), wx(5)
        for (int tt = 0; tt < 13; ++tt) {
            int l = map[tt];
            W[l] = (const float*)d_in[i++];
            Bv[l] = (const float*)d_in[i++];
        }
    }

    // --- workspace layout ---
    // Packed weights for convs 1..12 (wx0 is the direct kernel).
    size_t woff[13];
    size_t curHalves = 0;
    for (int l = 1; l < 13; ++l) {
        int Ktot = g_convs[l].Cin * g_convs[l].ks * g_convs[l].ks;   // % 32 == 0
        int Mpad = (g_convs[l].Cout + 15) & ~15;
        woff[l] = curHalves;
        curHalves += (size_t)Mpad * Ktot;
    }
    char* base = (char*)d_ws;
    _Float16* wpack = (_Float16*)base;
    size_t off = align256(curHalves * sizeof(_Float16));

    // f16 activation region (contiguous; zeroed once -> halos stay zero).
    _Float16* actBase = (_Float16*)(base + off);
    size_t acur = 0;
    auto alloc16 = [&](size_t elems) {
        _Float16* p = actBase + acur;
        acur += elems;
        return p;
    };
    _Float16* aP    = alloc16((size_t)BATCH * 130 * 130 * 1);
    _Float16* x1    = alloc16((size_t)BATCH * 130 * 130 * 32);
    _Float16* y32   = alloc16((size_t)BATCH * 66 * 66 * 32);
    _Float16* y64A  = alloc16((size_t)BATCH * 66 * 66 * 64);
    _Float16* y64B  = alloc16((size_t)BATCH * 66 * 66 * 64);
    _Float16* s128  = alloc16((size_t)BATCH * 66 * 66 * 128);
    _Float16* state = alloc16((size_t)BATCH * 66 * 66 * 128);
    _Float16* u64A  = alloc16((size_t)BATCH * 130 * 130 * 64);
    _Float16* u64B  = alloc16((size_t)BATCH * 130 * 130 * 64);
    _Float16* z128  = alloc16((size_t)BATCH * 130 * 130 * 128);
    _Float16* LG    = alloc16((size_t)BATCH * 16384 * 82);
    size_t actElems = acur;
    off += align256(actElems * sizeof(_Float16));

    float* a32  = (float*)(base + off); off += align256((size_t)BATCH * 16384 * 4);
    float* bufS = (float*)(base + off); off += align256((size_t)BATCH * 64 * 64 * 128 * 4);

    float* out = (float*)d_out;

    // --- pack weights (swizzled WMMA fragment layout, tap-major K) ---
    for (int l = 1; l < 13; ++l) {
        int KS2  = g_convs[l].ks * g_convs[l].ks;
        int Cin  = g_convs[l].Cin;
        int nCh  = (Cin * KS2) / 32;
        int Mpad = (g_convs[l].Cout + 15) & ~15;
        int n = (Mpad / 16) * nCh * 512;
        pack_weights_swz_kernel<<<(n + 255) / 256, 256, 0, stream>>>(
            W[l], wpack + woff[l], g_convs[l].Cout, Cin, KS2, nCh, n);
    }

    // --- zero all f16 activations (halos + initial state) ---
    {
        int n = (int)actElems;
        zero16_kernel<<<(n + 255) / 256, 256, 0, stream>>>(actBase, n);
    }

    const int NPIX = BATCH * 16384;
    for (int t = 0; t < TSTEPS; ++t) {
        blend_kernel<<<(NPIX + 255) / 256, 256, 0, stream>>>(
            frames, mask_true, out, a32, aP, t);

        // wx0 (Cin=1) direct
        conv1_direct_kernel<<<(NPIX + 255) / 256, 256, 0, stream>>>(
            aP, W[0], Bv[0], x1);

        // wx1: 32->32, stride2 @128
        conv_wmma_kernel<32, 3, 2, 128><<<dim3(16, 2, BATCH), 128, 0, stream>>>(
            x1, wpack + woff[1], Bv[1], nullptr, y32, nullptr, nullptr, 32, 1);
        // wx2: 32->64 @64
        conv_wmma_kernel<32, 3, 1, 64><<<dim3(16, 4, BATCH), 128, 0, stream>>>(
            y32, wpack + woff[2], Bv[2], nullptr, y64A, nullptr, nullptr, 64, 1);
        // wx3: 64->64 @64
        conv_wmma_kernel<64, 3, 1, 64><<<dim3(16, 4, BATCH), 128, 0, stream>>>(
            y64A, wpack + woff[3], Bv[3], nullptr, y64B, nullptr, nullptr, 64, 1);

        // ws0: 128->128 @64 (reads old state)
        conv_wmma_kernel<128, 3, 1, 64><<<dim3(16, 8, BATCH), 128, 0, stream>>>(
            state, wpack + woff[5], Bv[5], nullptr, s128, nullptr, nullptr, 128, 1);
        // ws1: 128->128 @64 -> f32 bufS
        conv_wmma_kernel<128, 3, 1, 64><<<dim3(16, 8, BATCH), 128, 0, stream>>>(
            s128, wpack + woff[6], Bv[6], nullptr, nullptr, nullptr, bufS, 128, 1);
        // wx4: 64->128 @64, + bufS -> new state
        conv_wmma_kernel<64, 3, 1, 64><<<dim3(16, 8, BATCH), 128, 0, stream>>>(
            y64B, wpack + woff[4], Bv[4], bufS, state, nullptr, nullptr, 128, 1);

        // wo0: 128->64 @64
        conv_wmma_kernel<128, 3, 1, 64><<<dim3(16, 4, BATCH), 128, 0, stream>>>(
            state, wpack + woff[7], Bv[7], nullptr, y64A, nullptr, nullptr, 64, 1);
        // wo1: 64->64 @64
        conv_wmma_kernel<64, 3, 1, 64><<<dim3(16, 4, BATCH), 128, 0, stream>>>(
            y64A, wpack + woff[8], Bv[8], nullptr, y64B, nullptr, nullptr, 64, 1);

        // upsample 2x (64ch)
        upsample2_kernel<<<((BATCH << 20) + 255) / 256, 256, 0, stream>>>(y64B, u64A);

        // wo2, wo3: 64->64 @128
        conv_wmma_kernel<64, 3, 1, 128><<<dim3(64, 4, BATCH), 128, 0, stream>>>(
            u64A, wpack + woff[9], Bv[9], nullptr, u64B, nullptr, nullptr, 64, 1);
        conv_wmma_kernel<64, 3, 1, 128><<<dim3(64, 4, BATCH), 128, 0, stream>>>(
            u64B, wpack + woff[10], Bv[10], nullptr, u64A, nullptr, nullptr, 64, 1);
        // wo4: 64->128 1x1 @128
        conv_wmma_kernel<64, 1, 1, 128><<<dim3(64, 8, BATCH), 128, 0, stream>>>(
            u64A, wpack + woff[11], Bv[11], nullptr, z128, nullptr, nullptr, 128, 1);
        // last: 128->82 1x1 @128, no relu -> unpadded NHWC logits
        conv_wmma_kernel<128, 1, 1, 128><<<dim3(64, 6, BATCH), 128, 0, stream>>>(
            z128, wpack + woff[12], Bv[12], nullptr, nullptr, LG, nullptr, 82, 0);

        dfn_kernel<<<(NPIX + 255) / 256, 256, 0, stream>>>(LG, a32, out, t);
    }
}
